// AIM_25280177504504
// MI455X (gfx1250) — matrix operand order repaired and checked by reading, not compiled, not run on hardware
//
#include <hip/hip_runtime.h>
#include <hip/hip_bf16.h>
#include <stdint.h>

#define NR   16384
#define OBS  1024
#define HID  2048
#define LAT  256
#define VOCAB 8192
#define HQ   4

typedef __bf16 bf16_t;
typedef __attribute__((ext_vector_type(16))) __bf16 v16bf;
typedef __attribute__((ext_vector_type(4)))  __bf16 v4bf;
typedef __attribute__((ext_vector_type(8)))  float  v8f;
typedef __attribute__((ext_vector_type(4)))  float  f32x4;
typedef unsigned int u32x4 __attribute__((ext_vector_type(4)));

union FragA { v16bf v; u32x4 q[2]; };
union FragC { v8f  v; float f[8]; };
struct ASet { FragA f0, f1, f2, f3; };   // named members: no dynamic indexing -> stays in VGPRs

// ---- CDNA5 async global->LDS copy (ASYNCcnt-tracked DMA path) ----
// LDS dest = low 32 bits of generic shared pointer.
__device__ __forceinline__ void async_ld_b128(const void* g, void* l) {
    asm volatile("global_load_async_to_lds_b128 %0, %1, off"
                 :: "v"((unsigned)(uintptr_t)l),
                    "v"((unsigned long long)(uintptr_t)g)
                 : "memory");
}
__device__ __forceinline__ void wait_async0() {
#if defined(__has_builtin) && __has_builtin(__builtin_amdgcn_s_wait_asynccnt)
    __builtin_amdgcn_s_wait_asynccnt(0);
#else
    asm volatile("s_wait_asynccnt 0x0" ::: "memory");
#endif
}

// ---------------- conversion kernels ----------------
__global__ void k_cvt(const float* __restrict__ s, bf16_t* __restrict__ d, long n) {
    long i = (long)blockIdx.x * blockDim.x + threadIdx.x;
    long stride = (long)gridDim.x * blockDim.x;
    for (; i < n; i += stride) d[i] = (bf16_t)s[i];
}

// s: [K][Nc] fp32 row-major  ->  d: [Nc][K] bf16 (transposed)
__global__ void k_cvt_t(const float* __restrict__ s, bf16_t* __restrict__ d, int K, int Nc) {
    __shared__ float tile[32][33];
    int bx = blockIdx.x * 32;   // Nc
    int by = blockIdx.y * 32;   // K
    int tx = threadIdx.x, ty = threadIdx.y;  // 32 x 8
    #pragma unroll
    for (int i = 0; i < 32; i += 8)
        tile[ty + i][tx] = s[(long)(by + ty + i) * Nc + bx + tx];
    __syncthreads();
    #pragma unroll
    for (int i = 0; i < 32; i += 8)
        d[(long)(bx + ty + i) * K + by + tx] = (bf16_t)tile[tx][ty + i];
}

// per-code squared norms, cb: [HQ*VOCAB][LAT]
__global__ void k_en(const float* __restrict__ cb, float* __restrict__ en) {
    int code = blockIdx.x * blockDim.x + threadIdx.x;
    const float* p = cb + (long)code * LAT;
    float s = 0.f;
    #pragma unroll 4
    for (int i = 0; i < LAT; i += 4) {
        f32x4 v = *(const f32x4*)(p + i);
        s += v[0]*v[0] + v[1]*v[1] + v[2]*v[2] + v[3]*v[3];
    }
    en[code] = s;
}

__global__ void k_zero(float* a) { a[0] = 0.f; a[1] = 0.f; }

__global__ void k_final(const float* __restrict__ a, float* __restrict__ out) {
    out[0] = 1.5f * a[0] / ((float)NR * (float)LAT)
           + 0.5f * a[1] / ((float)NR * (float)OBS);
}

// ---------------- WMMA GEMM (async-staged, double-buffered LDS) ----------------
// C[M,Ncol] = A[M,K](bf16) * Bt[Ncol,K](bf16)^T + bias
// EPI 0: store fp32   EPI 1: relu -> bf16   EPI 2: fused MSE vs Xref -> atomicAdd(accum)
template <int EPI>
__global__ __launch_bounds__(256, 1)
void k_gemm(const bf16_t* __restrict__ A, const bf16_t* __restrict__ Bt,
            const float* __restrict__ bias, float* __restrict__ Cf,
            bf16_t* __restrict__ Cb, const float* __restrict__ Xref,
            float* __restrict__ accum, int M, int K, int Ncol) {
    __shared__ bf16_t sA[2][128 * 48];
    __shared__ bf16_t sB[2][128 * 48];
    const int tid  = threadIdx.x;
    const int lane = tid & 31;
    const int w    = tid >> 5;
    const int rowBase = (w >> 2) * 64;   // 2 wave rows
    const int colBase = (w & 3) * 32;    // 4 wave cols
    const int lr = lane & 15;
    const int hi = lane >> 4;
    const long gM0 = (long)blockIdx.y * 128;
    const long gN0 = (long)blockIdx.x * 128;

    const int r0 = tid >> 1;                 // 0..127
    const int c0 = (tid & 1) * 16;           // 0 or 16

    auto stage = [&](int buf, int k0) {
        #pragma unroll
        for (int it = 0; it < 2; ++it) {
            int c8 = c0 + it * 8;
            async_ld_b128(A  + (gM0 + r0) * K + k0 + c8, &sA[buf][r0 * 48 + c8]);
            async_ld_b128(Bt + (gN0 + r0) * K + k0 + c8, &sB[buf][r0 * 48 + c8]);
        }
    };

    FragC acc[4][2];
    #pragma unroll
    for (int i = 0; i < 4; ++i)
        #pragma unroll
        for (int j = 0; j < 2; ++j)
            #pragma unroll
            for (int r = 0; r < 8; ++r) acc[i][j].f[r] = 0.f;

    stage(0, 0);
    int buf = 0;
    for (int k0 = 0; k0 < K; k0 += 32) {
        wait_async0();
        __syncthreads();
        if (k0 + 32 < K) stage(buf ^ 1, k0 + 32);

        const int oA = hi * 8;   // lane>=16 holds K 8..15 / 24..31
        const int oB = hi * 8;   // lane>=16 holds K 8..15 / 24..31
        FragA a0, a1, a2, a3, b0, b1;
        a0.q[0] = *(const u32x4*)(&sA[buf][(rowBase +  0 + lr) * 48 + oA]);
        a0.q[1] = *(const u32x4*)(&sA[buf][(rowBase +  0 + lr) * 48 + oA + 16]);
        a1.q[0] = *(const u32x4*)(&sA[buf][(rowBase + 16 + lr) * 48 + oA]);
        a1.q[1] = *(const u32x4*)(&sA[buf][(rowBase + 16 + lr) * 48 + oA + 16]);
        a2.q[0] = *(const u32x4*)(&sA[buf][(rowBase + 32 + lr) * 48 + oA]);
        a2.q[1] = *(const u32x4*)(&sA[buf][(rowBase + 32 + lr) * 48 + oA + 16]);
        a3.q[0] = *(const u32x4*)(&sA[buf][(rowBase + 48 + lr) * 48 + oA]);
        a3.q[1] = *(const u32x4*)(&sA[buf][(rowBase + 48 + lr) * 48 + oA + 16]);
        b0.q[0] = *(const u32x4*)(&sB[buf][(colBase +  0 + lr) * 48 + oB]);
        b0.q[1] = *(const u32x4*)(&sB[buf][(colBase +  0 + lr) * 48 + oB + 16]);
        b1.q[0] = *(const u32x4*)(&sB[buf][(colBase + 16 + lr) * 48 + oB]);
        b1.q[1] = *(const u32x4*)(&sB[buf][(colBase + 16 + lr) * 48 + oB + 16]);

        acc[0][0].v = __builtin_amdgcn_wmma_f32_16x16x32_bf16(false, a0.v, false, b0.v, (short)0, acc[0][0].v, false, false);
        acc[0][1].v = __builtin_amdgcn_wmma_f32_16x16x32_bf16(false, a0.v, false, b1.v, (short)0, acc[0][1].v, false, false);
        acc[1][0].v = __builtin_amdgcn_wmma_f32_16x16x32_bf16(false, a1.v, false, b0.v, (short)0, acc[1][0].v, false, false);
        acc[1][1].v = __builtin_amdgcn_wmma_f32_16x16x32_bf16(false, a1.v, false, b1.v, (short)0, acc[1][1].v, false, false);
        acc[2][0].v = __builtin_amdgcn_wmma_f32_16x16x32_bf16(false, a2.v, false, b0.v, (short)0, acc[2][0].v, false, false);
        acc[2][1].v = __builtin_amdgcn_wmma_f32_16x16x32_bf16(false, a2.v, false, b1.v, (short)0, acc[2][1].v, false, false);
        acc[3][0].v = __builtin_amdgcn_wmma_f32_16x16x32_bf16(false, a3.v, false, b0.v, (short)0, acc[3][0].v, false, false);
        acc[3][1].v = __builtin_amdgcn_wmma_f32_16x16x32_bf16(false, a3.v, false, b1.v, (short)0, acc[3][1].v, false, false);
        buf ^= 1;
    }

    float mse = 0.f;
    #pragma unroll
    for (int mf = 0; mf < 4; ++mf) {
        #pragma unroll
        for (int nf = 0; nf < 2; ++nf) {
            long col = gN0 + colBase + nf * 16 + lr;
            float bv = bias[col];
            #pragma unroll
            for (int r = 0; r < 8; ++r) {
                long row = gM0 + rowBase + mf * 16 + hi * 8 + r;
                float v = acc[mf][nf].f[r] + bv;
                if (EPI == 0) {
                    Cf[row * Ncol + col] = v;
                } else if (EPI == 1) {
                    Cb[row * Ncol + col] = (bf16_t)(v > 0.f ? v : 0.f);
                } else {
                    float d = v - Xref[row * Ncol + col];
                    mse += d * d;
                }
            }
        }
    }
    if (EPI == 2) atomicAdd(accum, mse);
}

// ---------------- LayerNorm + ReLU -> bf16 ----------------
__global__ __launch_bounds__(256)
void k_ln(const float* __restrict__ h, const float* __restrict__ g,
          const float* __restrict__ b, bf16_t* __restrict__ out) {
    __shared__ float red[256];
    const long row = blockIdx.x;
    const float* hp = h + row * HID;
    float lx[8];
    float s = 0.f;
    #pragma unroll
    for (int i = 0; i < 8; ++i) { lx[i] = hp[threadIdx.x + i * 256]; s += lx[i]; }
    red[threadIdx.x] = s;
    __syncthreads();
    for (int o = 128; o > 0; o >>= 1) {
        if (threadIdx.x < o) red[threadIdx.x] += red[threadIdx.x + o];
        __syncthreads();
    }
    float mu = red[0] / (float)HID;
    __syncthreads();
    float s2 = 0.f;
    #pragma unroll
    for (int i = 0; i < 8; ++i) { float d = lx[i] - mu; s2 += d * d; }
    red[threadIdx.x] = s2;
    __syncthreads();
    for (int o = 128; o > 0; o >>= 1) {
        if (threadIdx.x < o) red[threadIdx.x] += red[threadIdx.x + o];
        __syncthreads();
    }
    float rstd = rsqrtf(red[0] / (float)HID + 1e-5f);
    bf16_t* op = out + row * HID;
    #pragma unroll
    for (int i = 0; i < 8; ++i) {
        int c = threadIdx.x + i * 256;
        float v = (lx[i] - mu) * rstd * g[c] + b[c];
        op[c] = (bf16_t)(v > 0.f ? v : 0.f);
    }
}

// ---------------- fused residual VQ (4 levels) ----------------
__global__ __launch_bounds__(256, 1)
void k_rvq(const float* __restrict__ latent, const float* __restrict__ cb,
           const bf16_t* __restrict__ cbb, const float* __restrict__ en,
           float* __restrict__ resid, float* __restrict__ csum,
           bf16_t* __restrict__ quantb, float* __restrict__ accum) {
    __shared__ bf16_t sR[64 * 264];          // resid rows (stride 264: 16B aligned, bank-spread)
    __shared__ bf16_t sE[2][128 * 48];       // codebook chunk: 128 codes x 32 k, double buffered
    __shared__ unsigned long long mk[64];    // per-row packed min keys
    const int tid  = threadIdx.x;
    const int lane = tid & 31;
    const int w    = tid >> 5;     // wave -> 16-code column group
    const int lr   = lane & 15;
    const int hi   = lane >> 4;
    const long row0 = (long)blockIdx.x * 64;
    const int r0 = tid >> 1;
    const int c0 = (tid & 1) * 16;
    const int TOT = (VOCAB / 128) * (LAT / 32);   // 512 pipeline steps per level (even)

    // init resid = latent (fp32 in ws, bf16 in LDS), vectorized 16B
    for (int j = tid; j < 64 * LAT / 4; j += 256) {
        int r = j >> 6, c4 = (j & 63) * 4;
        f32x4 v = *(const f32x4*)(latent + (row0 + r) * LAT + c4);
        *(f32x4*)(resid + (row0 + r) * LAT + c4) = v;
        v4bf hv; hv[0] = (bf16_t)v[0]; hv[1] = (bf16_t)v[1];
        hv[2] = (bf16_t)v[2]; hv[3] = (bf16_t)v[3];
        *(v4bf*)(&sR[r * 264 + c4]) = hv;
    }
    __syncthreads();

    const int oA = hi * 8;
    const int oB = hi * 8;
    // sR is immutable within a level -> A fragments preloaded one k-step ahead
    // into the alternate named register set (static ping-pong, no spills).
    auto loadA = [&](ASet& a, int k0) {
        a.f0.q[0] = *(const u32x4*)(&sR[( 0 + lr) * 264 + k0 + oA]);
        a.f0.q[1] = *(const u32x4*)(&sR[( 0 + lr) * 264 + k0 + oA + 16]);
        a.f1.q[0] = *(const u32x4*)(&sR[(16 + lr) * 264 + k0 + oA]);
        a.f1.q[1] = *(const u32x4*)(&sR[(16 + lr) * 264 + k0 + oA + 16]);
        a.f2.q[0] = *(const u32x4*)(&sR[(32 + lr) * 264 + k0 + oA]);
        a.f2.q[1] = *(const u32x4*)(&sR[(32 + lr) * 264 + k0 + oA + 16]);
        a.f3.q[0] = *(const u32x4*)(&sR[(48 + lr) * 264 + k0 + oA]);
        a.f3.q[1] = *(const u32x4*)(&sR[(48 + lr) * 264 + k0 + oA + 16]);
    };

    float loss = 0.f;
    for (int l = 0; l < HQ; ++l) {
        if (tid < 64) mk[tid] = ~0ull;
        const bf16_t* Eb  = cbb + (long)l * VOCAB * LAT;
        const float*  enl = en  + (long)l * VOCAB;

        auto stage = [&](int buf, int s) {
            int ct = s >> 3, k0 = (s & 7) * 32;
            #pragma unroll
            for (int it = 0; it < 2; ++it) {
                int c8 = c0 + it * 8;
                async_ld_b128(Eb + (long)(ct * 128 + r0) * LAT + k0 + c8,
                              &sE[buf][r0 * 48 + c8]);
            }
        };

        FragC acc[4];
        #pragma unroll
        for (int mf = 0; mf < 4; ++mf)
            #pragma unroll
            for (int r = 0; r < 8; ++r) acc[mf].f[r] = 0.f;

        auto loadB = [&](FragA& b, int buf) {
            b.q[0] = *(const u32x4*)(&sE[buf][(w * 16 + lr) * 48 + oB]);
            b.q[1] = *(const u32x4*)(&sE[buf][(w * 16 + lr) * 48 + oB + 16]);
        };
        auto mul4 = [&](ASet& a, FragA& b) {
            acc[0].v = __builtin_amdgcn_wmma_f32_16x16x32_bf16(false, a.f0.v, false, b.v, (short)0, acc[0].v, false, false);
            acc[1].v = __builtin_amdgcn_wmma_f32_16x16x32_bf16(false, a.f1.v, false, b.v, (short)0, acc[1].v, false, false);
            acc[2].v = __builtin_amdgcn_wmma_f32_16x16x32_bf16(false, a.f2.v, false, b.v, (short)0, acc[2].v, false, false);
            acc[3].v = __builtin_amdgcn_wmma_f32_16x16x32_bf16(false, a.f3.v, false, b.v, (short)0, acc[3].v, false, false);
        };

        ASet aA, aB;
        loadA(aA, 0);                    // prologue: k-step 0 fragments
        stage(0, 0);
        int buf = 0;
        float e2 = 0.f;                  // prefetched ||E||^2 for this code tile
        // unrolled-by-2 stream: even half computes aA / preloads aB, odd half swaps.
        // e2 prefetch only fires at s%8==0 (even half); argmin epilogue only at
        // s%8==7 (odd half).
        for (int s = 0; s < TOT; s += 2) {
            // ---- even step: compute with aA, preload aB for k-step s+1
            wait_async0();
            __syncthreads();
            stage(buf ^ 1, s + 1);
            if ((s & 7) == 0)            // early load: ~8 iterations of latency cover
                e2 = enl[(s >> 3) * 128 + w * 16 + lr];
            FragA b0;
            loadB(b0, buf);
            loadA(aB, ((s + 1) & 7) * 32);
            mul4(aA, b0);
            buf ^= 1;

            // ---- odd step: compute with aB, preload aA for k-step s+2
            wait_async0();
            __syncthreads();
            if (s + 2 < TOT) stage(buf ^ 1, s + 2);
            FragA b1;
            loadB(b1, buf);
            loadA(aA, ((s + 2) & 7) * 32);
            mul4(aB, b1);

            if (((s + 1) & 7) == 7) {
                // score = ||E||^2 - 2*dot (row norm is argmin-invariant)
                int code = ((s + 1) >> 3) * 128 + w * 16 + lr;
                #pragma unroll
                for (int mf = 0; mf < 4; ++mf) {
                    #pragma unroll
                    for (int r = 0; r < 8; ++r) {
                        int row = mf * 16 + hi * 8 + r;
                        float sc = e2 - 2.f * acc[mf].f[r];
                        unsigned u = __float_as_uint(sc);
                        u = (u & 0x80000000u) ? ~u : (u | 0x80000000u);
                        unsigned long long key =
                            ((unsigned long long)u << 32) | (unsigned)code;
                        atomicMin(&mk[row], key);
                        acc[mf].f[r] = 0.f;
                    }
                }
            }
            buf ^= 1;
        }
        __syncthreads();   // all atomicMin visible

        // gather fp32 code, loss, residual update, bf16 refresh (vectorized 16B)
        const float* El = cb + (long)l * VOCAB * LAT;
        for (int j = tid; j < 64 * LAT / 4; j += 256) {
            int r = j >> 6, c4 = (j & 63) * 4;
            int code = (int)(mk[r] & 0xffffffffull);
            f32x4 q = *(const f32x4*)(El + (long)code * LAT + c4);
            long gi = (row0 + r) * LAT + c4;
            f32x4 rv = *(const f32x4*)(resid + gi);
            f32x4 d = q - rv;
            loss += d[0]*d[0] + d[1]*d[1] + d[2]*d[2] + d[3]*d[3];
            f32x4 nr = rv - q;
            *(f32x4*)(resid + gi) = nr;
            if (l == 0) {
                *(f32x4*)(csum + gi) = q;
            } else {
                f32x4 cs = *(const f32x4*)(csum + gi);
                *(f32x4*)(csum + gi) = cs + q;
            }
            v4bf hv; hv[0] = (bf16_t)nr[0]; hv[1] = (bf16_t)nr[1];
            hv[2] = (bf16_t)nr[2]; hv[3] = (bf16_t)nr[3];
            *(v4bf*)(&sR[r * 264 + c4]) = hv;
        }
        __syncthreads();
    }
    for (int j = tid; j < 64 * LAT / 4; j += 256) {
        int r = j >> 6, c4 = (j & 63) * 4;
        long gi = (row0 + r) * LAT + c4;
        f32x4 cs = *(const f32x4*)(csum + gi);
        v4bf hv; hv[0] = (bf16_t)cs[0]; hv[1] = (bf16_t)cs[1];
        hv[2] = (bf16_t)cs[2]; hv[3] = (bf16_t)cs[3];
        *(v4bf*)(&quantb[gi]) = hv;
    }
    atomicAdd(accum, loss);
}

// ---------------- launcher ----------------
extern "C" void kernel_launch(void* const* d_in, const int* in_sizes, int n_in,
                              void* d_out, int out_size, void* d_ws, size_t ws_size,
                              hipStream_t stream) {
    const float* x   = (const float*)d_in[0];
    const float* cb  = (const float*)d_in[1];
    const float* w1  = (const float*)d_in[2];
    const float* b1  = (const float*)d_in[3];
    const float* lg  = (const float*)d_in[4];
    const float* lb  = (const float*)d_in[5];
    const float* w2  = (const float*)d_in[6];
    const float* b2  = (const float*)d_in[7];
    const float* dw1 = (const float*)d_in[8];
    const float* db1 = (const float*)d_in[9];
    const float* dw2 = (const float*)d_in[10];
    const float* db2 = (const float*)d_in[11];
    float* out = (float*)d_out;
    (void)in_sizes; (void)n_in; (void)out_size; (void)ws_size;

    char* p = (char*)d_ws;
    auto alloc = [&](size_t bytes) {
        char* r = p;
        p += (bytes + 255) & ~(size_t)255;
        return (void*)r;
    };
    bf16_t* xb   = (bf16_t*)alloc((size_t)NR * OBS * 2);
    bf16_t* cbb  = (bf16_t*)alloc((size_t)HQ * VOCAB * LAT * 2);
    bf16_t* w1t  = (bf16_t*)alloc((size_t)OBS * HID * 2);
    bf16_t* w2t  = (bf16_t*)alloc((size_t)HID * LAT * 2);
    bf16_t* d1t  = (bf16_t*)alloc((size_t)LAT * HID * 2);
    bf16_t* d2t  = (bf16_t*)alloc((size_t)HID * OBS * 2);
    float*  en   = (float*)alloc((size_t)HQ * VOCAB * 4);
    float*  h    = (float*)alloc((size_t)NR * HID * 4);
    bf16_t* hb   = (bf16_t*)alloc((size_t)NR * HID * 2);
    float*  lat  = (float*)alloc((size_t)NR * LAT * 4);
    float*  rsd  = (float*)alloc((size_t)NR * LAT * 4);
    float*  csum = (float*)alloc((size_t)NR * LAT * 4);
    bf16_t* qb   = (bf16_t*)alloc((size_t)NR * LAT * 2);
    bf16_t* dhb  = (bf16_t*)alloc((size_t)NR * HID * 2);
    float*  acc  = (float*)alloc(2 * sizeof(float));

    k_zero<<<1, 1, 0, stream>>>(acc);
    k_cvt<<<1024, 256, 0, stream>>>(x,  xb,  (long)NR * OBS);
    k_cvt<<<1024, 256, 0, stream>>>(cb, cbb, (long)HQ * VOCAB * LAT);
    dim3 tb(32, 8);
    k_cvt_t<<<dim3(HID / 32, OBS / 32), tb, 0, stream>>>(w1,  w1t, OBS, HID);
    k_cvt_t<<<dim3(LAT / 32, HID / 32), tb, 0, stream>>>(w2,  w2t, HID, LAT);
    k_cvt_t<<<dim3(HID / 32, LAT / 32), tb, 0, stream>>>(dw1, d1t, LAT, HID);
    k_cvt_t<<<dim3(OBS / 32, HID / 32), tb, 0, stream>>>(dw2, d2t, HID, OBS);
    k_en<<<HQ * VOCAB / 256, 256, 0, stream>>>(cb, en);

    // encoder
    k_gemm<0><<<dim3(HID / 128, NR / 128), 256, 0, stream>>>(
        xb, w1t, b1, h, nullptr, nullptr, nullptr, NR, OBS, HID);
    k_ln<<<NR, 256, 0, stream>>>(h, lg, lb, hb);
    k_gemm<0><<<dim3(LAT / 128, NR / 128), 256, 0, stream>>>(
        hb, w2t, b2, lat, nullptr, nullptr, nullptr, NR, HID, LAT);

    // residual VQ (4 levels fused)
    k_rvq<<<NR / 64, 256, 0, stream>>>(lat, cb, cbb, en, rsd, csum, qb, acc);

    // decoder + fused reconstruction MSE
    k_gemm<1><<<dim3(HID / 128, NR / 128), 256, 0, stream>>>(
        qb, d1t, db1, nullptr, dhb, nullptr, nullptr, NR, LAT, HID);
    k_gemm<2><<<dim3(OBS / 128, NR / 128), 256, 0, stream>>>(
        dhb, d2t, db2, nullptr, nullptr, x, acc + 1, NR, HID, OBS);

    k_final<<<1, 1, 0, stream>>>(acc, out);
}
